// HyenaOperator2d_76355928588835
// MI455X (gfx1250) — matrix operand order repaired and checked by reading, not compiled, not run on hardware
//
#include <hip/hip_runtime.h>
#include <hip/hip_bf16.h>
#include <math.h>

// Problem dims
#define BB   4
#define SS   128
#define DD   256
#define D3   768
#define LL   (SS*SS)        // 16384
#define MROWS (BB*LL)       // 65536
#define FOW  64
#define NPAD (2*SS)         // fft pad length 256

typedef __attribute__((ext_vector_type(16))) __bf16 v16bf;
typedef __attribute__((ext_vector_type(8)))  float  v8f;
typedef unsigned int u32;

union FragBF { u32 u[8]; v16bf v; };

__device__ __forceinline__ v8f zero8() {
    v8f z = {0.f,0.f,0.f,0.f,0.f,0.f,0.f,0.f};
    return z;
}

// ---- WMMA fragment loads (wave32, 16x16x32 bf16) -------------------------
// A: 16x32 (MxK), row-major [M, lda]. Lane L: row = L&15, half = L>>4.
//   VGPR v in 0..7 : K = half*16 + 2v + {0,1}
__device__ __forceinline__ v16bf load_a_frag(const __bf16* __restrict__ A, int lda,
                                             int row0, int k0, int lane) {
    const int half = lane >> 4;
    const int r    = row0 + (lane & 15);
    const __bf16* p = A + (size_t)r * lda + k0;
    FragBF f;
#pragma unroll
    for (int v = 0; v < 8; ++v)
        f.u[v]     = *(const u32*)(p + half*16 + 2*v);
    return f.v;
}

// B: 32x16 (KxN) fragment out of an LDS-staged Bt tile [64 rows][Kld] bf16.
// Lane L: col n = L&15, half = L>>4. VGPR v: K = half*16 + 2v + {0,1}
// -> 32 contiguous bytes per lane (two ds_load_b128)
__device__ __forceinline__ v16bf load_bt_frag_lds(const __bf16* btile, int Kld,
                                                  int nt, int k0, int lane) {
    const int half = lane >> 4;
    const int n = nt * 16 + (lane & 15);
    const __bf16* p = btile + n * Kld + k0 + half * 16;
    FragBF f;
#pragma unroll
    for (int v = 0; v < 8; ++v)
        f.u[v] = *(const u32*)(p + 2 * v);
    return f.v;
}

// ---- async global -> LDS tile staging (gfx1250 ASYNC path) ---------------
// Copies `bytes` (multiple of 4096, 16B-aligned) from gsrc to LDS tile.
// blockDim.x must be 256. Issue and wait are split so callers can overlap
// independent global loads with the DMA.
__device__ __forceinline__ void async_stage_issue(const __bf16* __restrict__ gsrc,
                                                  __bf16* ldst, int bytes) {
    const u32 lbase = (u32)(uintptr_t)ldst;              // flat LDS addr[31:0] == LDS offset
    const char* g = (const char*)gsrc;
    for (int o = threadIdx.x * 16; o < bytes; o += 256 * 16) {
        u32 laddr = lbase + o;
        const char* ga = g + o;
        asm volatile("global_load_async_to_lds_b128 %0, %1, off"
                     :: "v"(laddr), "v"(ga)
                     : "memory");
    }
}
__device__ __forceinline__ void async_stage_wait() {
    asm volatile("s_wait_asynccnt 0x0" ::: "memory");
    __syncthreads();
}

// ---- Utility kernels -----------------------------------------------------
__global__ void k_cvt_bf16(const float* __restrict__ src, __bf16* __restrict__ dst, size_t n) {
    size_t i = (size_t)blockIdx.x * 256 + threadIdx.x;
    if (i < n) dst[i] = (__bf16)src[i];
}

// w[K,N] f32 -> wt[N,K] bf16
__global__ void k_wtrans(const float* __restrict__ w, __bf16* __restrict__ wt, int K, int N) {
    int idx = blockIdx.x * 256 + threadIdx.x;
    if (idx >= K * N) return;
    int n = idx % N, k = idx / N;
    wt[(size_t)n * K + k] = (__bf16)w[(size_t)k * N + n];
}

// ---- in_proj: u = x @ in_w + in_b  (M=65536, N=768, K=256) ---------------
// A kept resident in registers (8 k-step fragments); x streamed from HBM ONCE.
// B strips staged to LDS via async DMA, shared by all 8 waves.
__global__ void __launch_bounds__(256)
k_gemm_inproj(const __bf16* __restrict__ x, const __bf16* __restrict__ wt,
              const float* __restrict__ bias, __bf16* __restrict__ u) {
    __shared__ __bf16 btile[64 * DD];                    // 32 KB
    const int wav = threadIdx.x >> 5, lane = threadIdx.x & 31;
    const int row0 = blockIdx.x * 128 + wav * 16;
    v16bf areg[8];
#pragma unroll
    for (int ks = 0; ks < 8; ++ks)
        areg[ks] = load_a_frag(x, DD, row0, ks * 32, lane);
    const int n = lane & 15, half = lane >> 4;
    for (int strip = 0; strip < D3 / 64; ++strip) {
        const int col0 = strip * 64;
        __syncthreads();                                 // LDS reuse guard
        async_stage_issue(wt + (size_t)col0 * DD, btile, 64 * DD * 2);
        async_stage_wait();
        v8f c[4]; for (int i = 0; i < 4; ++i) c[i] = zero8();
#pragma unroll
        for (int ks = 0; ks < 8; ++ks) {
#pragma unroll
            for (int nt = 0; nt < 4; ++nt) {
                v16bf b = load_bt_frag_lds(btile, DD, nt, ks * 32, lane);
                c[nt] = __builtin_amdgcn_wmma_f32_16x16x32_bf16(
                    false, areg[ks], false, b, (short)0, c[nt], false, false);
            }
        }
#pragma unroll
        for (int nt = 0; nt < 4; ++nt) {
            const int col = col0 + nt*16 + n;
            const float bv = bias[col];
#pragma unroll
            for (int i = 0; i < 8; ++i) {
                const int m = row0 + i + 8*half;
                u[(size_t)m * D3 + col] = (__bf16)(c[nt][i] + bv);
            }
        }
    }
}

// ---- out_proj: out = Y @ out_w + out_b  (M=65536, N=256, K=256) ----------
__global__ void __launch_bounds__(256)
k_gemm_outproj(const __bf16* __restrict__ Y, const __bf16* __restrict__ wt,
               const float* __restrict__ bias, float* __restrict__ out) {
    __shared__ __bf16 btile[64 * DD];                    // 32 KB
    const int wav = threadIdx.x >> 5, lane = threadIdx.x & 31;
    const int row0 = blockIdx.x * 128 + wav * 16;
    v16bf areg[8];
#pragma unroll
    for (int ks = 0; ks < 8; ++ks)
        areg[ks] = load_a_frag(Y, DD, row0, ks * 32, lane);
    const int n = lane & 15, half = lane >> 4;
    for (int strip = 0; strip < DD / 64; ++strip) {
        const int col0 = strip * 64;
        __syncthreads();
        async_stage_issue(wt + (size_t)col0 * DD, btile, 64 * DD * 2);
        async_stage_wait();
        v8f c[4]; for (int i = 0; i < 4; ++i) c[i] = zero8();
#pragma unroll
        for (int ks = 0; ks < 8; ++ks) {
#pragma unroll
            for (int nt = 0; nt < 4; ++nt) {
                v16bf b = load_bt_frag_lds(btile, DD, nt, ks * 32, lane);
                c[nt] = __builtin_amdgcn_wmma_f32_16x16x32_bf16(
                    false, areg[ks], false, b, (short)0, c[nt], false, false);
            }
        }
#pragma unroll
        for (int nt = 0; nt < 4; ++nt) {
            const int col = col0 + nt*16 + n;
            const float bv = bias[col];
#pragma unroll
            for (int i = 0; i < 8; ++i) {
                const int m = row0 + i + 8*half;
                out[(size_t)m * DD + col] = c[nt][i] + bv;
            }
        }
    }
}

// ---- depthwise 3x3 conv (pad=2, crop :S) + split + gate ------------------
__global__ void k_dwconv(const __bf16* __restrict__ u, const float* __restrict__ sfw,
                         const float* __restrict__ sfb,
                         float* __restrict__ x0buf, __bf16* __restrict__ vg) {
    const int p = blockIdx.x;         // b*S*S + h*S + w
    const int d = threadIdx.x;        // channel within D
    const int b = p >> 14, rem = p & 16383;
    const int h = rem >> 7, w = rem & 127;
    float a0 = sfb[d], a1 = sfb[DD + d], a2 = sfb[2*DD + d];
#pragma unroll
    for (int i = 0; i < 3; ++i) {
        const int hh = h + i - 2;
        if (hh < 0) continue;
#pragma unroll
        for (int j = 0; j < 3; ++j) {
            const int ww = w + j - 2;
            if (ww < 0) continue;
            const size_t row = ((size_t)(b*SS + hh) * SS + ww) * D3;
            const int widx = i*3 + j;
            a0 += (float)u[row + d]        * sfw[(size_t)d        * 9 + widx];
            a1 += (float)u[row + DD + d]   * sfw[(size_t)(DD + d) * 9 + widx];
            a2 += (float)u[row + 2*DD + d] * sfw[(size_t)(2*DD+d) * 9 + widx];
        }
    }
    const size_t o = ((size_t)(b*DD + d) * SS + h) * SS + w;
    x0buf[o] = a0;                    // x0 (kept f32 for final gate)
    vg[o] = (__bf16)(a2 * a1);        // v * x1 (gated conv input)
}

// ---- implicit filter MLP + decay ----------------------------------------
__global__ void k_filter(const float* __restrict__ freq,
                         const float* xw0, const float* xb0, const float* xw1, const float* xb1,
                         const float* xw2, const float* xb2, const float* xw3,
                         const float* yw0, const float* yb0, const float* yw1, const float* yb1,
                         const float* yw2, const float* yb2, const float* yw3,
                         float* __restrict__ hx, float* __restrict__ hy) {
    __shared__ float sh[2][2][FOW];
    const int s = blockIdx.x, t = threadIdx.x;
    const float ts   = (float)s / (float)(SS - 1);
    const float wang = 6.28318530717958647692f * (float)s / (float)SS;
    const float f0   = 1e-4f;
    const float z0 = ts, z1 = cosf(f0 * wang), z2 = -sinf(f0 * wang);
    if (t < 2*FOW) {
        const int g = t >> 6, j = t & (FOW - 1);
        const float* w0 = g ? yw0 : xw0;
        const float* b0 = g ? yb0 : xb0;
        float a = z0*w0[j] + z1*w0[FOW + j] + z2*w0[2*FOW + j] + b0[j];
        sh[0][g][j] = sinf(freq[j] * a);
    }
    __syncthreads();
    if (t < 2*FOW) {
        const int g = t >> 6, j = t & (FOW - 1);
        const float* w = g ? yw1 : xw1;
        const float* bb = g ? yb1 : xb1;
        float a = bb[j];
        for (int k = 0; k < FOW; ++k) a += sh[0][g][k] * w[k*FOW + j];
        sh[1][g][j] = sinf(freq[j] * a);
    }
    __syncthreads();
    if (t < 2*FOW) {
        const int g = t >> 6, j = t & (FOW - 1);
        const float* w = g ? yw2 : xw2;
        const float* bb = g ? yb2 : xb2;
        float a = bb[j];
        for (int k = 0; k < FOW; ++k) a += sh[1][g][k] * w[k*FOW + j];
        sh[0][g][j] = sinf(freq[j] * a);
    }
    __syncthreads();
    const float dmin = logf(0.01f) / 1.5f;
    const float dmax = logf(0.01f) / 0.3f;
    const float delta = dmin + (dmax - dmin) * (float)t / (float)(DD - 1);
    const float dec = expf(-ts * fabsf(delta));
    float ax = 0.f, ay = 0.f;
    for (int k = 0; k < FOW; ++k) {
        ax += sh[0][0][k] * xw3[k*DD + t];
        ay += sh[0][1][k] * yw3[k*DD + t];
    }
    hx[(size_t)s * DD + t] = ax * dec * (1.0f / (float)NPAD);  // 1/(2S) of fft conv
    hy[(size_t)s * DD + t] = ay * dec;
}

// ---- build causal Toeplitz matrices (bf16) -------------------------------
// Tx[d][h][i] = hx[h-i, d] (h>=i);  TyT[d][w][j] = hy[w-j, d] (w>=j)
__global__ void k_toeplitz(const float* __restrict__ hx, const float* __restrict__ hy,
                           __bf16* __restrict__ tx, __bf16* __restrict__ tyT) {
    const size_t id = (size_t)blockIdx.x * 256 + threadIdx.x;   // over D*S*S
    const int i = (int)(id & 127);
    const size_t r = id >> 7;
    const int h = (int)(r & 127);
    const int d = (int)(r >> 7);
    const float vx = (h >= i) ? hx[(size_t)(h - i) * DD + d] : 0.f;
    const float vy = (h >= i) ? hy[(size_t)(h - i) * DD + d] : 0.f;
    tx[id]  = (__bf16)vx;
    tyT[id] = (__bf16)vy;
}

// ---- conv stage 1: Z[bd] = Vg[bd] @ Ty_d^T  (M=N=K=128 per (b,d)) --------
__global__ void __launch_bounds__(256)
k_gemm_conv_w(const __bf16* __restrict__ vg, const __bf16* __restrict__ tyT,
              __bf16* __restrict__ Z) {
    __shared__ __bf16 btile[64 * SS];                    // 16 KB
    const int bd = blockIdx.y;
    const int d  = bd & (DD - 1);
    const int wav = threadIdx.x >> 5, lane = threadIdx.x & 31;
    const int row0 = wav * 16;
    const int col0 = blockIdx.x * 64;
    const __bf16* A = vg + (size_t)bd * SS * SS;
    async_stage_issue(tyT + (size_t)d * SS * SS + (size_t)col0 * SS, btile, 64 * SS * 2);
    v16bf areg[4];
#pragma unroll
    for (int ks = 0; ks < 4; ++ks)
        areg[ks] = load_a_frag(A, SS, row0, ks * 32, lane);   // overlap with DMA
    async_stage_wait();
    v8f c[4]; for (int i = 0; i < 4; ++i) c[i] = zero8();
#pragma unroll
    for (int ks = 0; ks < 4; ++ks) {
#pragma unroll
        for (int nt = 0; nt < 4; ++nt) {
            v16bf b = load_bt_frag_lds(btile, SS, nt, ks * 32, lane);
            c[nt] = __builtin_amdgcn_wmma_f32_16x16x32_bf16(
                false, areg[ks], false, b, (short)0, c[nt], false, false);
        }
    }
    __bf16* out = Z + (size_t)bd * SS * SS;
    const int n = lane & 15, half = lane >> 4;
#pragma unroll
    for (int nt = 0; nt < 4; ++nt) {
        const int col = col0 + nt*16 + n;
#pragma unroll
        for (int i = 0; i < 8; ++i) {
            const int m = row0 + i + 8*half;
            out[(size_t)m * SS + col] = (__bf16)c[nt][i];
        }
    }
}

// ---- per-(b,d) 128x128 bf16 transpose via LDS ----------------------------
__global__ void k_transpose128(const __bf16* __restrict__ Z, __bf16* __restrict__ Zt) {
    __shared__ __bf16 tile[32][33];
    const int bd = blockIdx.z;
    const int x0 = blockIdx.x * 32, y0 = blockIdx.y * 32;
    const __bf16* src = Z  + (size_t)bd * SS * SS;
    __bf16*       dst = Zt + (size_t)bd * SS * SS;
    for (int r = threadIdx.y; r < 32; r += 8)
        tile[r][threadIdx.x] = src[(size_t)(y0 + r) * SS + x0 + threadIdx.x];
    __syncthreads();
    for (int r = threadIdx.y; r < 32; r += 8)
        dst[(size_t)(x0 + r) * SS + y0 + threadIdx.x] = tile[threadIdx.x][r];
}

// ---- conv stage 2 + epilogue: Y = (Tx_d @ Z + vg*fbias_d) * x0 -----------
// Stored re-laid-out to (B, L, D) bf16 for out_proj.
__global__ void __launch_bounds__(256)
k_gemm_conv_h(const __bf16* __restrict__ tx, const __bf16* __restrict__ Zt,
              const __bf16* __restrict__ vg, const float* __restrict__ x0buf,
              const float* __restrict__ fbias, __bf16* __restrict__ Y) {
    __shared__ __bf16 btile[64 * SS];                    // 16 KB
    const int bd = blockIdx.y;
    const int d  = bd & (DD - 1);
    const int b  = bd >> 8;
    const int wav = threadIdx.x >> 5, lane = threadIdx.x & 31;
    const int row0 = wav * 16;
    const int col0 = blockIdx.x * 64;
    const __bf16* A = tx + (size_t)d * SS * SS;
    async_stage_issue(Zt + (size_t)bd * SS * SS + (size_t)col0 * SS, btile, 64 * SS * 2);
    v16bf areg[4];
#pragma unroll
    for (int ks = 0; ks < 4; ++ks)
        areg[ks] = load_a_frag(A, SS, row0, ks * 32, lane);   // overlap with DMA
    async_stage_wait();
    v8f c[4]; for (int i = 0; i < 4; ++i) c[i] = zero8();
#pragma unroll
    for (int ks = 0; ks < 4; ++ks) {
#pragma unroll
        for (int nt = 0; nt < 4; ++nt) {
            v16bf bfrag = load_bt_frag_lds(btile, SS, nt, ks * 32, lane);
            c[nt] = __builtin_amdgcn_wmma_f32_16x16x32_bf16(
                false, areg[ks], false, bfrag, (short)0, c[nt], false, false);
        }
    }
    const float fb = fbias[d];
    const int n = lane & 15, half = lane >> 4;
#pragma unroll
    for (int nt = 0; nt < 4; ++nt) {
        const int w = col0 + nt*16 + n;
#pragma unroll
        for (int i = 0; i < 8; ++i) {
            const int h = row0 + i + 8*half;
            const size_t g = ((size_t)bd * SS + h) * SS + w;
            float val = c[nt][i] + (float)vg[g] * fb;   // conv + residual bias path
            val *= x0buf[g];                            // outer gate
            Y[((size_t)(b*SS + h) * SS + w) * DD + d] = (__bf16)val;
        }
    }
}

// ---- host launcher -------------------------------------------------------
static inline size_t alup(size_t x) { return (x + 255) & ~(size_t)255; }

extern "C" void kernel_launch(void* const* d_in, const int* in_sizes, int n_in,
                              void* d_out, int out_size, void* d_ws, size_t ws_size,
                              hipStream_t stream) {
    (void)in_sizes; (void)n_in; (void)out_size; (void)ws_size;
    const float* x    = (const float*)d_in[0];
    const float* in_w = (const float*)d_in[1];
    const float* in_b = (const float*)d_in[2];
    const float* out_w= (const float*)d_in[3];
    const float* out_b= (const float*)d_in[4];
    const float* sf_w = (const float*)d_in[5];
    const float* sf_b = (const float*)d_in[6];
    const float* freq = (const float*)d_in[7];
    const float* xw0 = (const float*)d_in[8];  const float* xb0 = (const float*)d_in[9];
    const float* xw1 = (const float*)d_in[10]; const float* xb1 = (const float*)d_in[11];
    const float* xw2 = (const float*)d_in[12]; const float* xb2 = (const float*)d_in[13];
    const float* xw3 = (const float*)d_in[14];
    const float* yw0 = (const float*)d_in[15]; const float* yb0 = (const float*)d_in[16];
    const float* yw1 = (const float*)d_in[17]; const float* yb1 = (const float*)d_in[18];
    const float* yw2 = (const float*)d_in[19]; const float* yb2 = (const float*)d_in[20];
    const float* yw3 = (const float*)d_in[21];
    const float* fbias = (const float*)d_in[22];
    float* out = (float*)d_out;

    // workspace carve-up
    char* base = (char*)d_ws;
    size_t off = 0;
    auto take = [&](size_t bytes) { char* p = base + off; off += alup(bytes); return p; };
    __bf16* xbf   = (__bf16*)take((size_t)MROWS * DD * 2);        // 33.6 MB
    __bf16* inwT  = (__bf16*)take((size_t)D3 * DD * 2);
    __bf16* outwT = (__bf16*)take((size_t)DD * DD * 2);
    __bf16* u     = (__bf16*)take((size_t)MROWS * D3 * 2);        // 100.7 MB
    float*  x0buf = (float*) take((size_t)BB * DD * SS * SS * 4); // 67.1 MB
    __bf16* vg    = (__bf16*)take((size_t)BB * DD * SS * SS * 2);
    float*  hx    = (float*) take((size_t)SS * DD * 4);
    float*  hy    = (float*) take((size_t)SS * DD * 4);
    __bf16* tx    = (__bf16*)take((size_t)DD * SS * SS * 2);
    __bf16* tyT   = (__bf16*)take((size_t)DD * SS * SS * 2);
    __bf16* Z     = (__bf16*)take((size_t)BB * DD * SS * SS * 2);
    __bf16* Zt    = (__bf16*)take((size_t)BB * DD * SS * SS * 2);
    __bf16* Y     = (__bf16*)take((size_t)MROWS * DD * 2);

    // 1. bf16 conversions / weight transposes
    k_cvt_bf16<<<(MROWS*DD + 255)/256, 256, 0, stream>>>(x, xbf, (size_t)MROWS * DD);
    k_wtrans<<<(DD*D3 + 255)/256, 256, 0, stream>>>(in_w, inwT, DD, D3);
    k_wtrans<<<(DD*DD + 255)/256, 256, 0, stream>>>(out_w, outwT, DD, DD);

    // 2. in_proj (WMMA, A-resident, async-LDS B staging)
    k_gemm_inproj<<<dim3(MROWS/128), 256, 0, stream>>>(xbf, inwT, in_b, u);

    // 3. depthwise 3x3 + split + gate
    k_dwconv<<<BB*SS*SS, DD, 0, stream>>>(u, sf_w, sf_b, x0buf, vg);

    // 4. implicit filter + Toeplitz
    k_filter<<<SS, 256, 0, stream>>>(freq, xw0, xb0, xw1, xb1, xw2, xb2, xw3,
                                     yw0, yb0, yw1, yb1, yw2, yb2, yw3, hx, hy);
    k_toeplitz<<<(DD*SS*SS)/256, 256, 0, stream>>>(hx, hy, tx, tyT);

    // 5. separable long conv as two WMMA GEMMs with a transpose between
    k_gemm_conv_w<<<dim3(SS/64, BB*DD), 256, 0, stream>>>(vg, tyT, Z);
    k_transpose128<<<dim3(SS/32, SS/32, BB*DD), dim3(32, 8), 0, stream>>>(Z, Zt);
    k_gemm_conv_h<<<dim3(SS/64, BB*DD), 256, 0, stream>>>(tx, Zt, vg, x0buf, fbias, Y);

    // 6. out_proj (WMMA, A-resident, async-LDS B staging)
    k_gemm_outproj<<<dim3(MROWS/128), 256, 0, stream>>>(Y, outwT, out_b, out);
}
